// Attention_82935818486035
// MI455X (gfx1250) — compile-verified, matrix operand order checked by reading, not run on hardware
//
#include <hip/hip_runtime.h>
#include <hip/hip_bf16.h>

// Problem dims (match reference)
#define BB 32
#define SS 2048
#define DHH 1024
#define PP_ 512
#define AA 512

typedef __attribute__((ext_vector_type(16))) __bf16        v16bf;
typedef __attribute__((ext_vector_type(16))) unsigned short v16u;
typedef __attribute__((ext_vector_type(8)))  float          v8f;
typedef __attribute__((ext_vector_type(4)))  float          v4f;
typedef __attribute__((ext_vector_type(4)))  unsigned int   v4u;

struct BPair { v4u lo; v4u hi; };   // 32 bytes, trivially copyable

__device__ __forceinline__ unsigned short f2bf(float f) {
    unsigned u = __builtin_bit_cast(unsigned, f);
    unsigned r = u + 0x7FFFu + ((u >> 16) & 1u);   // round-to-nearest-even
    return (unsigned short)(r >> 16);
}

// Branch-free tanh: v_exp_f32 (TRANS, co-executes with WMMA) + v_rcp_f32.
__device__ __forceinline__ float fast_tanh(float x) {
    float xc = fminf(fmaxf(x, -15.0f), 15.0f);
    float e  = __expf(2.0f * xc);
    return 1.0f - 2.0f * __builtin_amdgcn_rcpf(e + 1.0f);
}

// ---------------------------------------------------------------------------
// Kernel 1: WdT[n][k] = bf16(Wd[k][n])  (transpose + convert, one-time per call)
// ---------------------------------------------------------------------------
__global__ __launch_bounds__(256) void k_cvt_wd(const float* __restrict__ Wd,
                                               unsigned short* __restrict__ wdt) {
    int idx = blockIdx.x * 256 + threadIdx.x;          // 512*1024 total
    int n = idx >> 10;
    int k = idx & 1023;
    wdt[idx] = f2bf(Wd[(size_t)k * AA + n]);
}

// ---------------------------------------------------------------------------
// Kernel 2: pp[b][a] = bd[a] + sum_p pattern[b][p] * Wd[DH+p][a]
// ---------------------------------------------------------------------------
__global__ __launch_bounds__(256) void k_ppart(const float* __restrict__ pattern,
                                               const float* __restrict__ Wd,
                                               const float* __restrict__ bd,
                                               float* __restrict__ pp) {
    int idx = blockIdx.x * 256 + threadIdx.x;          // 32*512 total
    int b = idx >> 9;
    int a = idx & 511;
    float s = bd[a];
    const float* wp = Wd + (size_t)DHH * AA + a;
    const float* pb = pattern + (size_t)b * PP_;
    #pragma unroll 4
    for (int p = 0; p < PP_; ++p)
        s += pb[p] * wp[(size_t)p * AA];
    pp[idx] = s;
}

// ---------------------------------------------------------------------------
// Kernel 3: scores via WMMA bf16 GEMM + tanh + dot(Wv)
//   grid = (S/64, B), block = 512 threads = 16 waves
//   wave w: M-tile = w>>2 (16 rows of 64), N-chunk = (w&3)*128 (8 N-subtiles)
//   Wave-quads sharing an N-chunk reuse the same 8KB B K-slice via WGP$.
// ---------------------------------------------------------------------------
__global__ __launch_bounds__(512) void k_score(const float* __restrict__ hiddens,
                                               const unsigned short* __restrict__ wdt,
                                               const float* __restrict__ pp,
                                               const float* __restrict__ Wv,
                                               const float* __restrict__ bv,
                                               const unsigned char* __restrict__ mask,
                                               float* __restrict__ score) {
    const int b    = blockIdx.y;
    const int s0   = blockIdx.x * 64;
    const int tid  = threadIdx.x;
    const int wave = tid >> 5;
    const int lane = tid & 31;
    const int mtile = wave >> 2;           // 0..3
    const int n0    = (wave & 3) * 128;    // 0,128,256,384
    const int row   = lane & 15;
    const int khalf = (lane < 16) ? 0 : 8; // ISA 16-bit A-layout K split

    const float* hrow = hiddens + ((size_t)(b * SS + s0 + mtile * 16 + row)) * DHH;

    v8f acc[8];
    #pragma unroll
    for (int t = 0; t < 8; ++t)
        #pragma unroll
        for (int i = 0; i < 8; ++i) acc[t][i] = 0.0f;

    for (int kb = 0; kb < DHH; kb += 32) {
        // ---- A fragment: hiddens rows, fp32 -> bf16 on the fly ----
        const v4f* pa = (const v4f*)(hrow + kb);
        v4f fa0 = pa[(khalf >> 2) + 0];
        v4f fa1 = pa[(khalf >> 2) + 1];
        v4f fa2 = pa[(khalf >> 2) + 4];
        v4f fa3 = pa[(khalf >> 2) + 5];
        v16u au;
        #pragma unroll
        for (int i = 0; i < 4; ++i) {
            au[i]      = f2bf(fa0[i]);
            au[i + 4]  = f2bf(fa1[i]);
            au[i + 8]  = f2bf(fa2[i]);
            au[i + 12] = f2bf(fa3[i]);
        }
        v16bf afrag = __builtin_bit_cast(v16bf, au);

        // ---- 8 B fragments: WdT[n][k] bf16, aligned 16B chunks ----
        #pragma unroll
        for (int t = 0; t < 8; ++t) {
            int n = n0 + t * 16 + row;
            const unsigned short* wr = wdt + (size_t)n * DHH + kb + khalf;
            BPair bp;
            bp.lo = *(const v4u*)(wr);
            bp.hi = *(const v4u*)(wr + 16);
            v16bf bfrag = __builtin_bit_cast(v16bf, bp);
            acc[t] = __builtin_amdgcn_wmma_f32_16x16x32_bf16(
                false, afrag, false, bfrag, (short)0, acc[t], false, false);
        }
    }

    // ---- tanh, scale by Wv, per-row partial sums ----
    float partial[8];
    #pragma unroll
    for (int r = 0; r < 8; ++r) partial[r] = 0.0f;

    #pragma unroll
    for (int t = 0; t < 8; ++t) {
        int n = n0 + t * 16 + row;
        float ppn = pp[b * AA + n];
        float wvn = Wv[n];
        #pragma unroll
        for (int r = 0; r < 8; ++r)
            partial[r] += fast_tanh(acc[t][r] + ppn) * wvn;
    }

    // ---- reduce across the 16 lanes sharing each row (deterministic) ----
    #pragma unroll
    for (int m = 1; m < 16; m <<= 1)
        #pragma unroll
        for (int r = 0; r < 8; ++r)
            partial[r] += __shfl_xor(partial[r], m);

    __shared__ float sred[4][4][16];
    if ((lane & 15) == 0) {
        int base = (lane >> 4) * 8;   // lane0 -> rows 0..7, lane16 -> rows 8..15
        #pragma unroll
        for (int r = 0; r < 8; ++r)
            sred[mtile][wave & 3][base + r] = partial[r];
    }
    __syncthreads();

    if (tid < 64) {
        int mt = tid >> 4, m = tid & 15;
        float sum = bv[0];
        #pragma unroll
        for (int nc = 0; nc < 4; ++nc) sum += sred[mt][nc][m];   // fixed order
        int s = s0 + tid;
        if (!mask[(size_t)b * SS + s]) sum += -1.0e9f;
        score[(size_t)b * SS + s] = sum;
    }
}

// ---------------------------------------------------------------------------
// Kernel 4: softmax over S per batch (one block per b, deterministic trees)
// ---------------------------------------------------------------------------
__global__ __launch_bounds__(256) void k_softmax(const float* __restrict__ score,
                                                 float* __restrict__ attn) {
    int b = blockIdx.x, tid = threadIdx.x;
    __shared__ float red[256];
    float v[8];
    float m = -3.4e38f;
    #pragma unroll
    for (int i = 0; i < 8; ++i) {
        v[i] = score[(size_t)b * SS + i * 256 + tid];
        m = fmaxf(m, v[i]);
    }
    red[tid] = m; __syncthreads();
    for (int off = 128; off > 0; off >>= 1) {
        if (tid < off) red[tid] = fmaxf(red[tid], red[tid + off]);
        __syncthreads();
    }
    m = red[0]; __syncthreads();

    float e[8], s = 0.0f;
    #pragma unroll
    for (int i = 0; i < 8; ++i) { e[i] = __expf(v[i] - m); s += e[i]; }
    red[tid] = s; __syncthreads();
    for (int off = 128; off > 0; off >>= 1) {
        if (tid < off) red[tid] += red[tid + off];
        __syncthreads();
    }
    float inv = 1.0f / red[0];
    #pragma unroll
    for (int i = 0; i < 8; ++i)
        attn[(size_t)b * SS + i * 256 + tid] = e[i] * inv;
}

// ---------------------------------------------------------------------------
// Kernel 5: partial context per S-chunk: pctx[b][sc][d] = sum_{s in chunk} w*h
//   grid = (16, B), block = 256 (each thread: 4 consecutive d -> float4 loads)
// ---------------------------------------------------------------------------
__global__ __launch_bounds__(256) void k_ctx1(const float* __restrict__ hiddens,
                                              const float* __restrict__ attn,
                                              float* __restrict__ pctx) {
    int b = blockIdx.y, sc = blockIdx.x, tid = threadIdx.x;
    int d = tid * 4;
    const float* hb = hiddens + ((size_t)b * SS + sc * 128) * DHH + d;
    const float* ab = attn + (size_t)b * SS + sc * 128;
    v4f acc; acc[0] = acc[1] = acc[2] = acc[3] = 0.0f;
    for (int s = 0; s < 128; ++s) {
        __builtin_prefetch(hb + (size_t)(s + 4) * DHH, 0, 0);  // global_prefetch_b8
        float w = ab[s];
        v4f h = *(const v4f*)(hb + (size_t)s * DHH);
        acc[0] += w * h[0]; acc[1] += w * h[1];
        acc[2] += w * h[2]; acc[3] += w * h[3];
    }
    *(v4f*)(pctx + ((size_t)(b * 16 + sc)) * DHH + d) = acc;
}

// ---------------------------------------------------------------------------
// Kernel 6: reduce S-chunks in fixed order -> context[b][d]
// ---------------------------------------------------------------------------
__global__ __launch_bounds__(256) void k_ctx2(const float* __restrict__ pctx,
                                              float* __restrict__ out) {
    int b = blockIdx.x, d = threadIdx.x * 4;
    v4f acc; acc[0] = acc[1] = acc[2] = acc[3] = 0.0f;
    #pragma unroll
    for (int c = 0; c < 16; ++c) {
        v4f p = *(const v4f*)(pctx + ((size_t)(b * 16 + c)) * DHH + d);
        acc[0] += p[0]; acc[1] += p[1]; acc[2] += p[2]; acc[3] += p[3];
    }
    *(v4f*)(out + (size_t)b * DHH + d) = acc;
}

// ---------------------------------------------------------------------------
extern "C" void kernel_launch(void* const* d_in, const int* in_sizes, int n_in,
                              void* d_out, int out_size, void* d_ws, size_t ws_size,
                              hipStream_t stream) {
    (void)in_sizes; (void)n_in; (void)out_size; (void)ws_size;
    const float*         hiddens = (const float*)d_in[0];
    const float*         pattern = (const float*)d_in[1];
    const unsigned char* mask    = (const unsigned char*)d_in[2];
    const float*         Wd      = (const float*)d_in[3];
    const float*         bd      = (const float*)d_in[4];
    const float*         Wv      = (const float*)d_in[5];
    const float*         bv      = (const float*)d_in[6];
    float*               out     = (float*)d_out;

    // Workspace layout
    char* ws = (char*)d_ws;
    unsigned short* wdt  = (unsigned short*)(ws);                    // 1 MB
    float*          pp   = (float*)(ws + (1u << 20));                // 64 KB
    float*          scr  = (float*)(ws + (1u << 20) + (1u << 16));   // 256 KB
    float*          attn = (float*)(ws + (1u << 20) + (1u << 16) + (1u << 18)); // 256 KB
    float*          pctx = (float*)(ws + (1u << 20) + (1u << 16) + 2 * (1u << 18)); // 2 MB

    k_cvt_wd <<<dim3((AA * DHH) / 256), dim3(256), 0, stream>>>(Wd, wdt);
    k_ppart  <<<dim3((BB * AA) / 256),  dim3(256), 0, stream>>>(pattern, Wd, bd, pp);
    k_score  <<<dim3(SS / 64, BB),      dim3(512), 0, stream>>>(hiddens, wdt, pp, Wv, bv, mask, scr);
    k_softmax<<<dim3(BB),               dim3(256), 0, stream>>>(scr, attn);
    k_ctx1   <<<dim3(16, BB),           dim3(256), 0, stream>>>(hiddens, attn, pctx);
    k_ctx2   <<<dim3(BB),               dim3(256), 0, stream>>>(pctx, out);
}
